// EnergyTransformer_1786706395205
// MI455X (gfx1250) — matrix operand order repaired and checked by reading, not compiled, not run on hardware
//
#include <hip/hip_runtime.h>
#include <stdint.h>

// ---------------- problem constants ----------------
constexpr int   cB  = 8;
constexpr int   cN  = 1024;
constexpr int   cD  = 768;
constexpr int   cH  = 12;
constexpr int   cY  = 64;
constexpr int   cM  = 3072;
constexpr int   cBN = cB * cN;          // 8192 rows
constexpr int   cHY = cH * cY;          // 768
constexpr int   cK  = 2 * cHY + cM;     // 4608 (dK | dQ | hid)
constexpr int   cSTEPS = 12;
constexpr float kAlpha = 0.1f;
constexpr float kBeta  = 0.125f;        // 1/sqrt(64)
constexpr float kEps   = 1e-5f;

// LDS-staged B strip: 64 cols x 32 K bf16, row stride 80B (64B data + 16B pad)
constexpr int   cLdsRow = 40;           // elements per row (80 B)

// ---------------- WMMA plumbing ----------------
typedef __attribute__((ext_vector_type(16))) __bf16 v16bf;
typedef __attribute__((ext_vector_type(8)))  float  v8f;
typedef __attribute__((ext_vector_type(4)))  unsigned int v4u;
typedef __attribute__((ext_vector_type(8)))  int    v8i;
typedef __attribute__((ext_vector_type(4)))  int    v4i;

union ABF { uint32_t u[8]; v16bf v; };

__device__ __forceinline__ unsigned short f2bf(float f) {
  union { float f; uint32_t u; } x; x.f = f;
  uint32_t r = x.u + 0x7FFFu + ((x.u >> 16) & 1u);   // round-to-nearest-even
  return (unsigned short)(r >> 16);
}

__device__ __forceinline__ v8f wmma_bf(v16bf a, v16bf b, v8f c) {
  return __builtin_amdgcn_wmma_f32_16x16x32_bf16(false, a, false, b, (short)0, c, false, false);
}

// A operand (16 rows x 32 K) from row-major [row][k] bf16.
__device__ __forceinline__ v16bf ld_tA(const unsigned short* p, int ld, int r0, int k0, int lane) {
  const unsigned short* q = p + (size_t)(r0 + (lane & 15)) * ld + k0 + ((lane & 16) ? 8 : 0);
  uint4 a = *(const uint4*)q;
  uint4 b = *(const uint4*)(q + 16);
  ABF t;
  t.u[0]=a.x; t.u[1]=a.y; t.u[2]=a.z; t.u[3]=a.w;
  t.u[4]=b.x; t.u[5]=b.y; t.u[6]=b.z; t.u[7]=b.w;
  return t.v;
}

// B operand (32 K x 16 cols) from [col][k] row-major bf16 in global memory.
__device__ __forceinline__ v16bf ld_tB(const unsigned short* p, int ld, int c0, int k0, int lane) {
  const unsigned short* q = p + (size_t)(c0 + (lane & 15)) * ld + k0 + ((lane & 16) ? 8 : 0);
  uint4 a = *(const uint4*)q;
  uint4 b = *(const uint4*)(q + 16);
  ABF t;
  t.u[0]=a.x; t.u[1]=a.y; t.u[2]=a.z; t.u[3]=a.w;
  t.u[4]=b.x; t.u[5]=b.y; t.u[6]=b.z; t.u[7]=b.w;
  return t.v;
}

// B operand from TDM-staged LDS strip: [64 cols][32 k], padded row stride 40 elems.
__device__ __forceinline__ v16bf ld_tB_lds(const unsigned short* base, int c0, int lane) {
  const unsigned short* q = base + (size_t)(c0 + (lane & 15)) * cLdsRow + ((lane & 16) ? 8 : 0);
  uint4 a = *(const uint4*)q;
  uint4 b = *(const uint4*)(q + 16);
  ABF t;
  t.u[0]=a.x; t.u[1]=a.y; t.u[2]=a.z; t.u[3]=a.w;
  t.u[4]=b.x; t.u[5]=b.y; t.u[6]=b.z; t.u[7]=b.w;
  return t.v;
}

// ---------------- Tensor Data Mover: 2D tile -> LDS ----------------
// Builds a D# per CDNA5 ISA (group0 / group1 bit layout).
// Tile: tile1 rows (cols of the GEMM B strip) x tile0 contiguous elements,
// data_size = 2 bytes, LDS pad: 4 DWORDs after every 16 DWORDs (80B row stride).
__device__ __forceinline__ void tdm_load_2d(unsigned lds_off, const void* gaddr,
                                            unsigned dim0, unsigned dim1,
                                            unsigned tile0, unsigned tile1,
                                            unsigned stride0) {
  unsigned long long ga = (unsigned long long)(uintptr_t)gaddr;
  v4u g0;
  g0.x = 1u;                                        // count=1, user descriptor
  g0.y = lds_off;                                   // LDS byte address
  g0.z = (unsigned)(ga & 0xFFFFFFFFu);              // global_addr[31:0]
  g0.w = (unsigned)((ga >> 32) & 0x01FFFFFFu) | (2u << 30);  // addr[56:32] | type=2
  v8i g1;
  // data_size=1 (2B) | pad_enable | pad_interval=3 (16 DW) | pad_amount=3 (4 DW)
  g1[0] = (int)((1u << 16) | (1u << 20) | (3u << 22) | (3u << 25));
  g1[1] = (int)((dim0 & 0xFFFFu) << 16);            // tensor_dim0[15:0]
  g1[2] = (int)((dim0 >> 16) | ((dim1 & 0xFFFFu) << 16));
  g1[3] = (int)((dim1 >> 16) | (tile0 << 16));      // tensor_dim1[31:16] | tile_dim0
  g1[4] = (int)(tile1 & 0xFFFFu);                   // tile_dim1 | tile_dim2=0
  g1[5] = (int)stride0;                             // tensor_dim0_stride[31:0]
  g1[6] = 0;                                        // stride0[47:32] | stride1[15:0]
  g1[7] = 0;
  v4i z4 = {0, 0, 0, 0};
#if __clang_major__ >= 23
  v8i z8 = {0, 0, 0, 0, 0, 0, 0, 0};
  __builtin_amdgcn_tensor_load_to_lds(g0, g1, z4, z4, z8, 0);
#else
  __builtin_amdgcn_tensor_load_to_lds(g0, g1, z4, z4, 0);
#endif
}

// ---------------- prep kernels (run once per launch) ----------------
__global__ void k_castw(const float* __restrict__ wq, const float* __restrict__ wk,
                        unsigned short* __restrict__ wqb, unsigned short* __restrict__ wkb) {
  int i = blockIdx.x * blockDim.x + threadIdx.x;
  if (i < cHY * cD) { wqb[i] = f2bf(wq[i]); wkb[i] = f2bf(wk[i]); }
}

__global__ void k_castxi(const float* __restrict__ xi, unsigned short* __restrict__ xib) {
  int i = blockIdx.x * blockDim.x + threadIdx.x;
  if (i < cM * cD) xib[i] = f2bf(xi[i]);
}

// BigW[d][k] = Wk[k,d] | Wq[k-768,d] | xi[k-1536,d]   (768 x 4608)
__global__ void k_bigw(const float* __restrict__ wq, const float* __restrict__ wk,
                       const float* __restrict__ xi, unsigned short* __restrict__ bw) {
  int i = blockIdx.x * blockDim.x + threadIdx.x;
  if (i >= cD * cK) return;
  int d = i / cK, k = i % cK;
  float v;
  if (k < cHY)            v = wk[(size_t)k * cD + d];
  else if (k < 2 * cHY)   v = wq[(size_t)(k - cHY) * cD + d];
  else                    v = xi[(size_t)(k - 2 * cHY) * cD + d];
  bw[i] = f2bf(v);
}

// ---------------- per-step kernels ----------------
// LayerNorm: x[row] -> g_bf16[row]
__global__ void k_ln(const float* __restrict__ x, const float* __restrict__ gamma,
                     const float* __restrict__ beta, unsigned short* __restrict__ g) {
  int row = blockIdx.x;
  const float* xr = x + (size_t)row * cD;
  float s = 0.f, s2 = 0.f;
  for (int i = threadIdx.x; i < cD; i += blockDim.x) { float v = xr[i]; s += v; s2 += v * v; }
  for (int m = 16; m >= 1; m >>= 1) { s += __shfl_xor(s, m, 32); s2 += __shfl_xor(s2, m, 32); }
  __shared__ float ssum[8], ssq[8], smu, srv;
  int wid = threadIdx.x >> 5;
  if ((threadIdx.x & 31) == 0) { ssum[wid] = s; ssq[wid] = s2; }
  __syncthreads();
  if (threadIdx.x == 0) {
    float a = 0.f, b = 0.f;
    for (int i = 0; i < 8; i++) { a += ssum[i]; b += ssq[i]; }
    float mu = a / cD;
    smu = mu; srv = rsqrtf(b / cD - mu * mu + kEps);
  }
  __syncthreads();
  float mu = smu, rv = srv;
  for (int i = threadIdx.x; i < cD; i += blockDim.x)
    g[(size_t)row * cD + i] = f2bf((xr[i] - mu) * rv * gamma[i] + beta[i]);
}

// Q2/K2 = g @ W^T with TDM-staged, double-buffered weight strips in LDS.
__global__ void k_proj(const unsigned short* __restrict__ g,
                       const unsigned short* __restrict__ wqb, const unsigned short* __restrict__ wkb,
                       unsigned short* __restrict__ Q2, unsigned short* __restrict__ K2,
                       unsigned short* __restrict__ QT, unsigned short* __restrict__ KT) {
  __shared__ __align__(16) unsigned short sbq[2][64 * cLdsRow];
  __shared__ __align__(16) unsigned short sbk[2][64 * cLdsRow];
  int wid = threadIdx.x >> 5, lane = threadIdx.x & 31;
  int r0 = blockIdx.y * 128 + wid * 16;   // bn row tile
  int c0 = blockIdx.x * 64;               // hy col strip (shared by all 8 waves)
  if (wid == 0) {
    tdm_load_2d((unsigned)(uintptr_t)&sbq[0][0], wqb + (size_t)c0 * cD, cD, cHY, 32, 64, cD);
    tdm_load_2d((unsigned)(uintptr_t)&sbk[0][0], wkb + (size_t)c0 * cD, cD, cHY, 32, 64, cD);
  }
  v8f z = {};
  v8f aq[4] = {z, z, z, z}, ak[4] = {z, z, z, z};
  for (int i = 0; i < cD / 32; i++) {
    int k0 = i * 32, cur = i & 1, nxt = cur ^ 1;
    if (wid == 0) {
      if (k0 + 32 < cD) {
        tdm_load_2d((unsigned)(uintptr_t)&sbq[nxt][0], wqb + (size_t)c0 * cD + k0 + 32, cD, cHY, 32, 64, cD);
        tdm_load_2d((unsigned)(uintptr_t)&sbk[nxt][0], wkb + (size_t)c0 * cD + k0 + 32, cD, cHY, 32, 64, cD);
        __builtin_amdgcn_s_wait_tensorcnt(2);
      } else {
        __builtin_amdgcn_s_wait_tensorcnt(0);
      }
    }
    __syncthreads();
    v16bf a = ld_tA(g, cD, r0, k0, lane);
    for (int t = 0; t < 4; t++) {
      aq[t] = wmma_bf(a, ld_tB_lds(sbq[cur], 16 * t, lane), aq[t]);
      ak[t] = wmma_bf(a, ld_tB_lds(sbk[cur], 16 * t, lane), ak[t]);
    }
    __syncthreads();
  }
  int half = (lane & 16) ? 8 : 0, cl = lane & 15;
  for (int t = 0; t < 4; t++)
    for (int v = 0; v < 8; v++) {
      int row = r0 + v + half;            // bn
      int hy  = c0 + 16 * t + cl;
      int b = row >> 10, n = row & 1023, h = hy >> 6, y = hy & 63;
      unsigned short qv = f2bf(aq[t][v]), kv = f2bf(ak[t][v]);
      Q2[(size_t)row * cHY + hy] = qv;
      K2[(size_t)row * cHY + hy] = kv;
      size_t to = ((size_t)(b * cH + h) * cY + y) * cN + n;
      QT[to] = qv; KT[to] = kv;
    }
}

// lse[b,h,n] = logsumexp_m( beta * <K_m, Q_n> )  — online softmax over key tiles.
__global__ void k_lse(const unsigned short* __restrict__ Q2, const unsigned short* __restrict__ K2,
                      float* __restrict__ lse) {
  int wid = threadIdx.x >> 5, lane = threadIdx.x & 31;
  int bh = blockIdx.x, b = bh / cH, h = bh % cH;
  int n0 = blockIdx.y * 128 + wid * 16;
  const unsigned short* qb = Q2 + (size_t)b * cN * cHY + h * cY;
  const unsigned short* kb = K2 + (size_t)b * cN * cHY + h * cY;
  v16bf a0 = ld_tA(qb, cHY, n0, 0, lane);
  v16bf a1 = ld_tA(qb, cHY, n0, 32, lane);
  float mr[8], sr[8];
  for (int v = 0; v < 8; v++) { mr[v] = -1e30f; sr[v] = 0.f; }
  for (int m0 = 0; m0 < cN; m0 += 16) {
    v8f c = {};
    c = wmma_bf(a0, ld_tB(kb, cHY, m0, 0, lane), c);
    c = wmma_bf(a1, ld_tB(kb, cHY, m0, 32, lane), c);
    for (int v = 0; v < 8; v++) {
      float t = kBeta * c[v];
      float mn = fmaxf(mr[v], t);
      sr[v] = sr[v] * __expf(mr[v] - mn) + __expf(t - mn);
      mr[v] = mn;
    }
  }
  for (int msk = 1; msk < 16; msk <<= 1)     // reduce across the 16-lane half
    for (int v = 0; v < 8; v++) {
      float mo = __shfl_xor(mr[v], msk, 32);
      float so = __shfl_xor(sr[v], msk, 32);
      float mn = fmaxf(mr[v], mo);
      sr[v] = sr[v] * __expf(mr[v] - mn) + so * __expf(mo - mn);
      mr[v] = mn;
    }
  if ((lane & 15) == 0) {
    int half = (lane & 16) ? 8 : 0;
    for (int v = 0; v < 8; v++)
      lse[(size_t)bh * cN + n0 + v + half] = mr[v] + __logf(sr[v]);
  }
}

// dQ = P^T K  -> BigA columns [768, 1536)
__global__ void k_dq(const unsigned short* __restrict__ Q2, const unsigned short* __restrict__ K2,
                     const unsigned short* __restrict__ KT, const float* __restrict__ lse,
                     unsigned short* __restrict__ BigA) {
  __shared__ __align__(16) unsigned short pb[8][512];   // 16 x 32 per wave
  int wid = threadIdx.x >> 5, lane = threadIdx.x & 31;
  int bh = blockIdx.x, b = bh / cH, h = bh % cH;
  int n0 = blockIdx.y * 128 + wid * 16;
  const unsigned short* qb  = Q2 + (size_t)b * cN * cHY + h * cY;
  const unsigned short* kb  = K2 + (size_t)b * cN * cHY + h * cY;
  const unsigned short* ktb = KT + (size_t)bh * cY * cN;
  v16bf a0 = ld_tA(qb, cHY, n0, 0, lane);
  v16bf a1 = ld_tA(qb, cHY, n0, 32, lane);
  int half = (lane & 16) ? 8 : 0, cl = lane & 15;
  float l[8];
  for (int v = 0; v < 8; v++) l[v] = lse[(size_t)bh * cN + n0 + v + half];
  v8f z = {};
  v8f dq[4] = {z, z, z, z};
  unsigned short* myp = pb[wid];
  for (int m0 = 0; m0 < cN; m0 += 32) {
    for (int t = 0; t < 2; t++) {                 // sT[n, m] tiles
      v8f c = {};
      c = wmma_bf(a0, ld_tB(kb, cHY, m0 + 16 * t, 0, lane), c);
      c = wmma_bf(a1, ld_tB(kb, cHY, m0 + 16 * t, 32, lane), c);
      for (int v = 0; v < 8; v++)
        myp[(v + half) * 32 + 16 * t + cl] = f2bf(__expf(kBeta * c[v] - l[v]));
    }
    __syncthreads();
    v16bf pa;                                     // re-read as A operand (16n x 32m)
    {
      const unsigned short* q = myp + (size_t)(lane & 15) * 32 + ((lane & 16) ? 8 : 0);
      uint4 x0 = *(const uint4*)q;
      uint4 x1 = *(const uint4*)(q + 16);
      ABF t; t.u[0]=x0.x; t.u[1]=x0.y; t.u[2]=x0.z; t.u[3]=x0.w;
             t.u[4]=x1.x; t.u[5]=x1.y; t.u[6]=x1.z; t.u[7]=x1.w;
      pa = t.v;
    }
    for (int t = 0; t < 4; t++)
      dq[t] = wmma_bf(pa, ld_tB(ktb, cN, 16 * t, m0, lane), dq[t]);
    __syncthreads();
  }
  for (int t = 0; t < 4; t++)
    for (int v = 0; v < 8; v++) {
      int row = b * cN + n0 + v + half;
      int col = cHY + h * cY + 16 * t + cl;
      BigA[(size_t)row * cK + col] = f2bf(dq[t][v]);
    }
}

// dK = P Q  -> BigA columns [0, 768)
__global__ void k_dk(const unsigned short* __restrict__ Q2, const unsigned short* __restrict__ K2,
                     const unsigned short* __restrict__ QT, const float* __restrict__ lse,
                     unsigned short* __restrict__ BigA) {
  __shared__ __align__(16) unsigned short pb[8][512];
  int wid = threadIdx.x >> 5, lane = threadIdx.x & 31;
  int bh = blockIdx.x, b = bh / cH, h = bh % cH;
  int m0 = blockIdx.y * 128 + wid * 16;
  const unsigned short* qb  = Q2 + (size_t)b * cN * cHY + h * cY;
  const unsigned short* kb  = K2 + (size_t)b * cN * cHY + h * cY;
  const unsigned short* qtb = QT + (size_t)bh * cY * cN;
  v16bf a0 = ld_tA(kb, cHY, m0, 0, lane);
  v16bf a1 = ld_tA(kb, cHY, m0, 32, lane);
  int half = (lane & 16) ? 8 : 0, cl = lane & 15;
  v8f z = {};
  v8f dk[4] = {z, z, z, z};
  unsigned short* myp = pb[wid];
  for (int n0 = 0; n0 < cN; n0 += 32) {
    for (int t = 0; t < 2; t++) {                 // s[m, n] tiles; lse indexed by col n
      v8f c = {};
      c = wmma_bf(a0, ld_tB(qb, cHY, n0 + 16 * t, 0, lane), c);
      c = wmma_bf(a1, ld_tB(qb, cHY, n0 + 16 * t, 32, lane), c);
      float ln = lse[(size_t)bh * cN + n0 + 16 * t + cl];
      for (int v = 0; v < 8; v++)
        myp[(v + half) * 32 + 16 * t + cl] = f2bf(__expf(kBeta * c[v] - ln));
    }
    __syncthreads();
    v16bf pa;
    {
      const unsigned short* q = myp + (size_t)(lane & 15) * 32 + ((lane & 16) ? 8 : 0);
      uint4 x0 = *(const uint4*)q;
      uint4 x1 = *(const uint4*)(q + 16);
      ABF t; t.u[0]=x0.x; t.u[1]=x0.y; t.u[2]=x0.z; t.u[3]=x0.w;
             t.u[4]=x1.x; t.u[5]=x1.y; t.u[6]=x1.z; t.u[7]=x1.w;
      pa = t.v;
    }
    for (int t = 0; t < 4; t++)
      dk[t] = wmma_bf(pa, ld_tB(qtb, cN, 16 * t, n0, lane), dk[t]);
    __syncthreads();
  }
  for (int t = 0; t < 4; t++)
    for (int v = 0; v < 8; v++) {
      int row = b * cN + m0 + v + half;
      int col = h * cY + 16 * t + cl;
      BigA[(size_t)row * cK + col] = f2bf(dk[t][v]);
    }
}

// relu(g @ xi^T) -> BigA columns [1536, 4608); xi strip TDM-staged in LDS.
__global__ void k_hid(const unsigned short* __restrict__ g, const unsigned short* __restrict__ xib,
                      unsigned short* __restrict__ BigA) {
  __shared__ __align__(16) unsigned short sbx[2][64 * cLdsRow];
  int wid = threadIdx.x >> 5, lane = threadIdx.x & 31;
  int r0 = blockIdx.y * 128 + wid * 16;
  int c0 = blockIdx.x * 64;               // m strip (shared by all waves)
  if (wid == 0)
    tdm_load_2d((unsigned)(uintptr_t)&sbx[0][0], xib + (size_t)c0 * cD, cD, cM, 32, 64, cD);
  v8f z = {};
  v8f acc[4] = {z, z, z, z};
  for (int i = 0; i < cD / 32; i++) {
    int k0 = i * 32, cur = i & 1, nxt = cur ^ 1;
    if (wid == 0) {
      if (k0 + 32 < cD) {
        tdm_load_2d((unsigned)(uintptr_t)&sbx[nxt][0], xib + (size_t)c0 * cD + k0 + 32, cD, cM, 32, 64, cD);
        __builtin_amdgcn_s_wait_tensorcnt(1);
      } else {
        __builtin_amdgcn_s_wait_tensorcnt(0);
      }
    }
    __syncthreads();
    v16bf a = ld_tA(g, cD, r0, k0, lane);
    for (int t = 0; t < 4; t++)
      acc[t] = wmma_bf(a, ld_tB_lds(sbx[cur], 16 * t, lane), acc[t]);
    __syncthreads();
  }
  int half = (lane & 16) ? 8 : 0, cl = lane & 15;
  for (int t = 0; t < 4; t++)
    for (int v = 0; v < 8; v++)
      BigA[(size_t)(r0 + v + half) * cK + 2 * cHY + c0 + 16 * t + cl] = f2bf(fmaxf(acc[t][v], 0.f));
}

// Fused gradient + update: x += alpha * (BigA @ BigW^T); BigW strip TDM-staged.
__global__ void k_grad(const unsigned short* __restrict__ BigA, const unsigned short* __restrict__ BigW,
                       float* __restrict__ xbuf) {
  __shared__ __align__(16) unsigned short sbw[2][64 * cLdsRow];
  int wid = threadIdx.x >> 5, lane = threadIdx.x & 31;
  int r0 = blockIdx.y * 128 + wid * 16;
  int d0 = blockIdx.x * 64;
  if (wid == 0)
    tdm_load_2d((unsigned)(uintptr_t)&sbw[0][0], BigW + (size_t)d0 * cK, cK, cD, 32, 64, cK);
  v8f z = {};
  v8f acc[4] = {z, z, z, z};
  for (int i = 0; i < cK / 32; i++) {
    int k0 = i * 32, cur = i & 1, nxt = cur ^ 1;
    if (wid == 0) {
      if (k0 + 32 < cK) {
        tdm_load_2d((unsigned)(uintptr_t)&sbw[nxt][0], BigW + (size_t)d0 * cK + k0 + 32, cK, cD, 32, 64, cK);
        __builtin_amdgcn_s_wait_tensorcnt(1);
      } else {
        __builtin_amdgcn_s_wait_tensorcnt(0);
      }
    }
    __syncthreads();
    v16bf a = ld_tA(BigA, cK, r0, k0, lane);
    for (int t = 0; t < 4; t++)
      acc[t] = wmma_bf(a, ld_tB_lds(sbw[cur], 16 * t, lane), acc[t]);
    __syncthreads();
  }
  int half = (lane & 16) ? 8 : 0, cl = lane & 15;
  for (int t = 0; t < 4; t++)
    for (int v = 0; v < 8; v++) {
      size_t idx = (size_t)(r0 + v + half) * cD + d0 + 16 * t + cl;
      xbuf[idx] += kAlpha * acc[t][v];
    }
}

// ---------------- launch ----------------
extern "C" void kernel_launch(void* const* d_in, const int* in_sizes, int n_in,
                              void* d_out, int out_size, void* d_ws, size_t ws_size,
                              hipStream_t stream) {
  const float* x     = (const float*)d_in[0];
  const float* gamma = (const float*)d_in[1];
  const float* beta  = (const float*)d_in[2];
  const float* wq    = (const float*)d_in[3];
  const float* wk    = (const float*)d_in[4];
  const float* xi    = (const float*)d_in[5];

  char* w = (char*)d_ws;
  auto alloc = [&](size_t bytes) -> char* {
    char* p = w; w += (bytes + 255) & ~(size_t)255; return p;
  };
  float*          xbuf = (float*)         alloc(sizeof(float) * cBN * cD);
  unsigned short* gbf  = (unsigned short*)alloc(2ull * cBN * cD);
  unsigned short* Q2   = (unsigned short*)alloc(2ull * cBN * cHY);
  unsigned short* K2   = (unsigned short*)alloc(2ull * cBN * cHY);
  unsigned short* QT   = (unsigned short*)alloc(2ull * cB * cH * cY * cN);
  unsigned short* KT   = (unsigned short*)alloc(2ull * cB * cH * cY * cN);
  float*          lseb = (float*)         alloc(sizeof(float) * cB * cH * cN);
  unsigned short* BigA = (unsigned short*)alloc(2ull * cBN * cK);
  unsigned short* BigW = (unsigned short*)alloc(2ull * cD * cK);
  unsigned short* wqb  = (unsigned short*)alloc(2ull * cHY * cD);
  unsigned short* wkb  = (unsigned short*)alloc(2ull * cHY * cD);
  unsigned short* xib  = (unsigned short*)alloc(2ull * cM * cD);

  hipMemcpyAsync(xbuf, x, sizeof(float) * cBN * cD, hipMemcpyDeviceToDevice, stream);

  k_castw <<<(cHY * cD + 255) / 256, 256, 0, stream>>>(wq, wk, wqb, wkb);
  k_castxi<<<(cM * cD + 255) / 256, 256, 0, stream>>>(xi, xib);
  k_bigw  <<<(cD * cK + 255) / 256, 256, 0, stream>>>(wq, wk, xi, BigW);

  for (int s = 0; s < cSTEPS; s++) {
    k_ln  <<<cBN, 256, 0, stream>>>(xbuf, gamma, beta, gbf);
    k_proj<<<dim3(cHY / 64, cBN / 128), 256, 0, stream>>>(gbf, wqb, wkb, Q2, K2, QT, KT);
    k_lse <<<dim3(cB * cH, cN / 128), 256, 0, stream>>>(Q2, K2, lseb);
    k_dq  <<<dim3(cB * cH, cN / 128), 256, 0, stream>>>(Q2, K2, KT, lseb, BigA);
    k_dk  <<<dim3(cB * cH, cN / 128), 256, 0, stream>>>(Q2, K2, QT, lseb, BigA);
    k_hid <<<dim3(cM / 64, cBN / 128), 256, 0, stream>>>(gbf, xib, BigA);
    k_grad<<<dim3(cD / 64, cBN / 128), 256, 0, stream>>>(BigA, BigW, xbuf);
  }

  hipMemcpyAsync(d_out, xbuf, sizeof(float) * cBN * cD, hipMemcpyDeviceToDevice, stream);
}
